// WGNN_GAT_78847009620175
// MI455X (gfx1250) — matrix operand order repaired and checked by reading, not compiled, not run on hardware
//
#include <hip/hip_runtime.h>
#include <hip/hip_bf16.h>

typedef __attribute__((ext_vector_type(16))) _Float16 v16h;
typedef __attribute__((ext_vector_type(8)))  float    v8f;

#define IN_C  128
#define HID_C 96
#define OUT_C 64
#define ATT_SLOPE 0.2f
#define OUT_SLOPE 0.01f

static inline long cdiv(long a, long b) { return (a + b - 1) / b; }

// ---------------------------------------------------------------------------
// Generic fill (smax init, denom zero, accumulator zero)
// ---------------------------------------------------------------------------
__global__ void fill_u32_kernel(unsigned* __restrict__ p, unsigned v, long n) {
    long i = (long)blockIdx.x * blockDim.x + threadIdx.x;
    if (i < n) p[i] = v;
}

// ---------------------------------------------------------------------------
// WMMA GEMM: xl = x @ Wl + bl ; xr = x @ Wr + br
// 256 threads = 8 waves; each wave owns 4 consecutive 16x16 M-tiles in one
// 16-wide output-channel column. B fragments (both weight matrices, all K
// steps) are preloaded into registers once per wave and reused across the 4
// tiles. A rows are clamped (not predicated) so all loads are straight-line
// float2 -> v_cvt_pk_f16_f32 -> v_wmma_f32_16x16x32_f16.
// A/B use contiguous k = 16*half + s (same in both); D layout per CDNA5 ISA §7.12.2 (wave32).
// ---------------------------------------------------------------------------
template <int CIN, int COUT>
__global__ __launch_bounds__(256)
void gat_gemm_wmma_kernel(const float* __restrict__ x,   // [n, CIN]
                          const float* __restrict__ Wl,  // [CIN, COUT]
                          const float* __restrict__ Wr,  // [CIN, COUT]
                          const float* __restrict__ bl,  // [COUT]
                          const float* __restrict__ br,  // [COUT]
                          float* __restrict__ xl,        // [n, COUT]
                          float* __restrict__ xr,        // [n, COUT]
                          int n_nodes)
{
    constexpr int KSTEPS = CIN / 32;
    const int lane = threadIdx.x & 31;
    const int wave = threadIdx.x >> 5;
    const int half = lane >> 4;        // 0: lanes 0-15, 1: lanes 16-31
    const int sub  = lane & 15;
    const int n0   = blockIdx.y * 16;  // out-channel tile
    const int col  = n0 + sub;

    // ---- Preload B fragments: lanes 0-15 hold K=0..15, lanes 16-31 K=16..31 ----
    v16h bL[KSTEPS], bR[KSTEPS];
#pragma unroll
    for (int ks = 0; ks < KSTEPS; ++ks) {
#pragma unroll
        for (int v = 0; v < 8; ++v) {
            int k = ks * 32 + half * 16 + 2 * v;
            bL[ks][2 * v]     = (_Float16)Wl[k * COUT + col];
            bL[ks][2 * v + 1] = (_Float16)Wl[(k + 1) * COUT + col];
            bR[ks][2 * v]     = (_Float16)Wr[k * COUT + col];
            bR[ks][2 * v + 1] = (_Float16)Wr[(k + 1) * COUT + col];
        }
    }
    const float addL = bl[col];
    const float addR = br[col];

#pragma unroll
    for (int t = 0; t < 4; ++t) {
        const int mtile = (blockIdx.x * 8 + wave) * 4 + t;
        const int m0 = mtile * 16;
        if (m0 >= n_nodes) return;               // wave-uniform exit
        int arow = m0 + sub;
        if (arow >= n_nodes) arow = n_nodes - 1; // clamp: keep loads unpredicated
        const float2* xrow = reinterpret_cast<const float2*>(x + (long)arow * CIN);

        v8f cl = {};
        v8f cr = {};
#pragma unroll
        for (int ks = 0; ks < KSTEPS; ++ks) {
            // A fragment: VGPR v : K = half*16 + 2v
            v16h a;
#pragma unroll
            for (int v = 0; v < 8; ++v) {
                int k = ks * 32 + half * 16 + 2 * v;
                float2 f = xrow[k >> 1];
                a[2 * v]     = (_Float16)f.x;
                a[2 * v + 1] = (_Float16)f.y;
            }
            cl = __builtin_amdgcn_wmma_f32_16x16x32_f16(false, a, false, bL[ks], (short)0, cl, false, false);
            cr = __builtin_amdgcn_wmma_f32_16x16x32_f16(false, a, false, bR[ks], (short)0, cr, false, false);
        }
        // D layout: VGPR v holds row m0 + v + 8*half, column = col
#pragma unroll
        for (int v = 0; v < 8; ++v) {
            int r = m0 + v + 8 * half;
            if (r < n_nodes) {
                xl[(long)r * COUT + col] = cl[v] + addL;
                xr[(long)r * COUT + col] = cr[v] + addR;
            }
        }
    }
}

// ---------------------------------------------------------------------------
// Ordered-uint encoding for float atomic max (monotone mapping)
// ---------------------------------------------------------------------------
__device__ __forceinline__ unsigned float_to_ord(float f) {
    unsigned u = __float_as_uint(f);
    return (u & 0x80000000u) ? ~u : (u | 0x80000000u);
}
__device__ __forceinline__ float ord_to_float(unsigned o) {
    unsigned u = (o & 0x80000000u) ? (o ^ 0x80000000u) : ~o;
    return __uint_as_float(u);
}

// ---------------------------------------------------------------------------
// Edge logits: one wave per edge; channel loop fully unrolled (COUT/32 iters).
// s[e] = sum_c att[c] * leaky_relu(xl[src][c] + xr[dst][c] + ew[e]*We[c], 0.2)
// ---------------------------------------------------------------------------
template <int COUT>
__global__ __launch_bounds__(256)
void edge_logits_kernel(const float* __restrict__ xl,
                        const float* __restrict__ xr,
                        const float* __restrict__ ew,   // [E]
                        const float* __restrict__ We,   // [COUT]
                        const float* __restrict__ att,  // [COUT]
                        const int* __restrict__ src,
                        const int* __restrict__ dst,
                        float* __restrict__ s,          // [E]
                        unsigned* __restrict__ smax,    // [N] ordered
                        int n_edges)
{
    const int wid  = blockIdx.x * (blockDim.x >> 5) + (threadIdx.x >> 5);
    const int lane = threadIdx.x & 31;
    if (wid >= n_edges) return;
    const int u = src[wid], v = dst[wid];
    const float w = ew[wid];
    float acc = 0.0f;
#pragma unroll
    for (int i = 0; i < COUT / 32; ++i) {
        const int c = lane + i * 32;
        float m = xl[(long)u * COUT + c] + xr[(long)v * COUT + c] + w * We[c];
        m = (m > 0.0f) ? m : ATT_SLOPE * m;
        acc += m * att[c];
    }
#pragma unroll
    for (int off = 16; off > 0; off >>= 1)
        acc += __shfl_xor(acc, off, 32);
    if (lane == 0) {
        s[wid] = acc;
        atomicMax(&smax[v], float_to_ord(acc));
    }
}

// ---------------------------------------------------------------------------
// Edge exp: ex[e] = exp(s[e] - smax[dst]); denom[dst] += ex   (in-place on s)
// ---------------------------------------------------------------------------
__global__ __launch_bounds__(256)
void edge_exp_kernel(float* __restrict__ s,              // [E] in/out
                     const unsigned* __restrict__ smax,  // [N]
                     const int* __restrict__ dst,
                     float* __restrict__ denom,          // [N]
                     int n_edges)
{
    const int e = blockIdx.x * blockDim.x + threadIdx.x;
    if (e >= n_edges) return;
    const int v = dst[e];
    const float val = __expf(s[e] - ord_to_float(smax[v]));
    s[e] = val;
    atomicAdd(&denom[v], val);
}

// ---------------------------------------------------------------------------
// Edge aggregate: one wave per edge.
// out[dst][c] += (ex[e]/denom[dst]) * xl[src][c]
// ---------------------------------------------------------------------------
template <int COUT>
__global__ __launch_bounds__(256)
void edge_aggr_kernel(const float* __restrict__ ex,      // [E]
                      const float* __restrict__ denom,   // [N]
                      const float* __restrict__ xl,
                      const int* __restrict__ src,
                      const int* __restrict__ dst,
                      float* __restrict__ out,           // [N, COUT]
                      int n_edges)
{
    const int wid  = blockIdx.x * (blockDim.x >> 5) + (threadIdx.x >> 5);
    const int lane = threadIdx.x & 31;
    if (wid >= n_edges) return;
    const int u = src[wid], v = dst[wid];
    const float alpha = ex[wid] / denom[v];
#pragma unroll
    for (int i = 0; i < COUT / 32; ++i) {
        const int c = lane + i * 32;
        atomicAdd(&out[(long)v * COUT + c], alpha * xl[(long)u * COUT + c]);
    }
}

// ---------------------------------------------------------------------------
// Finalize: y = activation(acc + b)   (leaky slope; 0 => ReLU)
// ---------------------------------------------------------------------------
template <int COUT>
__global__ __launch_bounds__(256)
void finalize_kernel(const float* __restrict__ acc,
                     const float* __restrict__ b,
                     float* __restrict__ y,
                     long total, float slope)
{
    const long i = (long)blockIdx.x * blockDim.x + threadIdx.x;
    if (i >= total) return;
    const float val = acc[i] + b[(int)(i % COUT)];
    y[i] = (val > 0.0f) ? val : slope * val;
}

// ---------------------------------------------------------------------------
// Host side
// ---------------------------------------------------------------------------
struct LayerP {
    const float *We, *Wl, *Wr, *att, *b, *bl, *br;
};

template <int CIN, int COUT>
static void run_layer(const float* in, const LayerP& P, float* acc, float* dest,
                      float slope, int N, int E, const int* src, const int* dst,
                      const float* ew, float* xl, float* xr, float* sbuf,
                      float* denom, unsigned* smax, hipStream_t stream)
{
    const long nc = (long)N * COUT;
    // init segment buffers + accumulator
    fill_u32_kernel<<<cdiv(N, 256), 256, 0, stream>>>(smax, 0u, N);
    fill_u32_kernel<<<cdiv(N, 256), 256, 0, stream>>>((unsigned*)denom, 0u, N);
    fill_u32_kernel<<<cdiv(nc, 256), 256, 0, stream>>>((unsigned*)acc, 0u, nc);
    // dense projections via WMMA (8 waves x 4 M-tiles = 512 rows per block)
    dim3 ggrid((unsigned)cdiv(N, 512), (unsigned)(COUT / 16));
    gat_gemm_wmma_kernel<CIN, COUT><<<ggrid, 256, 0, stream>>>(
        in, P.Wl, P.Wr, P.bl, P.br, xl, xr, N);
    // edge phase (wave per edge, 8 waves per block)
    const long ewb = cdiv(E, 8);
    edge_logits_kernel<COUT><<<ewb, 256, 0, stream>>>(
        xl, xr, ew, P.We, P.att, src, dst, sbuf, smax, E);
    edge_exp_kernel<<<cdiv(E, 256), 256, 0, stream>>>(sbuf, smax, dst, denom, E);
    edge_aggr_kernel<COUT><<<ewb, 256, 0, stream>>>(
        sbuf, denom, xl, src, dst, acc, E);
    // bias + activation
    finalize_kernel<COUT><<<cdiv(nc, 256), 256, 0, stream>>>(acc, P.b, dest, nc, slope);
}

extern "C" void kernel_launch(void* const* d_in, const int* in_sizes, int n_in,
                              void* d_out, int out_size, void* d_ws, size_t ws_size,
                              hipStream_t stream) {
    const float* x  = (const float*)d_in[0];
    const int*   ei = (const int*)d_in[1];
    const float* ew = (const float*)d_in[2];

    const int N = in_sizes[0] / IN_C;   // 50000
    const int E = in_sizes[2];          // 800000 (EDGE_D == 1)

    const int* src = ei;
    const int* dst = ei + E;

    // JAX pytree dict flattening: sorted keys -> We, Wl, Wr, att, b, bl, br
    auto getP = [&](int base) {
        LayerP p;
        p.We  = (const float*)d_in[base + 0];
        p.Wl  = (const float*)d_in[base + 1];
        p.Wr  = (const float*)d_in[base + 2];
        p.att = (const float*)d_in[base + 3];
        p.b   = (const float*)d_in[base + 4];
        p.bl  = (const float*)d_in[base + 5];
        p.br  = (const float*)d_in[base + 6];
        return p;
    };
    const LayerP P1 = getP(3), P2 = getP(10), P3 = getP(17);

    // Workspace carve-up (floats): 4*N*96 + E + 2*N  ~ 81 MB
    float* xl    = (float*)d_ws;             // N * 96
    float* xr    = xl + (long)N * HID_C;     // N * 96
    float* y1    = xr + (long)N * HID_C;     // N * 96  (layer1 out / layer3 accum)
    float* y2    = y1 + (long)N * HID_C;     // N * 96  (layer2 out)
    float* sbuf  = y2 + (long)N * HID_C;     // E
    float* denom = sbuf + E;                 // N
    unsigned* smax = (unsigned*)(denom + N); // N (ordered uints)

    float* out = (float*)d_out;
    run_layer<IN_C,  HID_C>(x,  P1, y1, y1,  0.0f,      N, E, src, dst, ew, xl, xr, sbuf, denom, smax, stream); // ReLU
    run_layer<HID_C, HID_C>(y1, P2, y2, y2,  0.0f,      N, E, src, dst, ew, xl, xr, sbuf, denom, smax, stream); // ReLU
    run_layer<HID_C, OUT_C>(y2, P3, y1, out, OUT_SLOPE, N, E, src, dst, ew, xl, xr, sbuf, denom, smax, stream); // leaky 0.01
}
